// TransformerBlock_13477607375144
// MI455X (gfx1250) — matrix operand order repaired and checked by reading, not compiled, not run on hardware
//
#include <hip/hip_runtime.h>
#include <hip/hip_bf16.h>
#include <stdint.h>

// ---------------------------------------------------------------------------
// MI455X (gfx1250) transformer block: f16 WMMA (f32 accumulate) everywhere.
// ~55 GFLOP vs ~200MB @ 23.3 TB/s -> matrix-core bound. Weights are
// transpose-converted once so all LDS fragment reads are contiguous b128
// (no v_mov_b16 fixup). GEMM B tiles and attention K tiles are staged with
// GLOBAL_LOAD_ASYNC_TO_LDS_B128 (ASYNCcnt); softmax runs in exp2 domain.
// ---------------------------------------------------------------------------

typedef _Float16 f16_t;
typedef __attribute__((ext_vector_type(16))) _Float16 v16h;
typedef __attribute__((ext_vector_type(8)))  float    v8f;

union FragH { v16h v; uint32_t u[8]; _Float16 h[16]; };
union PackH { uint32_t u; _Float16 h[2]; };
union Pack4H { uint2 q; _Float16 h[4]; };

// A-fragment K index for 16x32 f16 A matrix (contiguous order, matches B frags):
// lanes 0-15 hold K 0-15 (VGPR0-7); lanes 16-31: K 16-31.
__device__ __forceinline__ int a_k_of(int i, int hi) {
  return hi * 16 + 2 * i;
}

__device__ __forceinline__ void async_copy_b128(uint32_t lds_off, uint64_t gaddr) {
  asm volatile("global_load_async_to_lds_b128 %0, %1, off"
               :: "v"(lds_off), "v"(gaddr)
               : "memory");
}
__device__ __forceinline__ void wait_asynccnt0() {
  asm volatile("s_wait_asynccnt 0x0" ::: "memory");
}

// ---------------------------------------------------------------------------
// GEMM: [8192 x 512] (f32 A, cvt on LDS stage) @ Wt[512 x 512] (f16, PRE-
// TRANSPOSED [n][k]) + bias.
// MODE 0: f16 out, [B,T,H*dh] -> [B,H,T,dh] head permute (QKV projections)
// MODE 1: f32 out, row-major              MODE 2: f32 out with leaky_relu
// Block tile 128x64, BK=32, 256 threads = 8 waves, each wave a 32x32 tile.
// B tile async-DMA'd global->LDS; both frag types read as merged b128.
// ---------------------------------------------------------------------------
template <int MODE>
__global__ __launch_bounds__(256) void gemm512(
    const float* __restrict__ A, const _Float16* __restrict__ Wt,
    const float* __restrict__ bias, _Float16* __restrict__ out16,
    float* __restrict__ out32) {
  __shared__ __align__(16) _Float16 As[128][40];  // [m][k], 80B stride (b128 ok)
  __shared__ __align__(16) _Float16 Bs[64][40];   // [n][k], async-copied

  const int tid   = threadIdx.x;
  const int lane  = tid & 31;
  const int w     = tid >> 5;
  const int waveM = w >> 1, waveN = w & 1;
  const int m16   = lane & 15, hi = lane >> 4;
  const int blockM = blockIdx.y, blockN = blockIdx.x;

  v8f zero = {};
  v8f acc[2][2];
#pragma unroll
  for (int a = 0; a < 2; ++a)
#pragma unroll
    for (int b = 0; b < 2; ++b) acc[a][b] = zero;

  const int am  = tid >> 1;
  const int ac0 = (tid & 1) * 16;
  const float* aSrcRow = A + (size_t)(blockM * 128 + am) * 512 + ac0;
  // B async staging: 64 n-rows x 32 k-halves (64B/row); 16B per thread
  const int bn   = tid >> 2;          // 0..63 : n row
  const int bkof = (tid & 3) * 8;     // 0,8,16,24 halves
  const _Float16* bSrc = Wt + (size_t)(blockN * 64 + bn) * 512 + bkof;
  const uint32_t bLds = (uint32_t)(uintptr_t)&Bs[bn][bkof];

  for (int kk = 0; kk < 512; kk += 32) {
    // --- B tile: async DMA global -> LDS (overlaps A staging below) ---
    async_copy_b128(bLds, (uint64_t)(uintptr_t)(bSrc + kk));
    // --- A tile: f32 -> f16 on the fly through VGPRs, packed b64 stores ---
    const float4* s4 = reinterpret_cast<const float4*>(aSrcRow + kk);
#pragma unroll
    for (int j = 0; j < 4; ++j) {
      float4 f = s4[j];
      Pack4H p;
      p.h[0] = (_Float16)f.x; p.h[1] = (_Float16)f.y;
      p.h[2] = (_Float16)f.z; p.h[3] = (_Float16)f.w;
      *reinterpret_cast<uint2*>(&As[am][ac0 + j * 4]) = p.q;
    }
    __builtin_prefetch(aSrcRow + kk + 32, 0, 1);  // next A k-slice
    wait_asynccnt0();
    __syncthreads();

    FragH aF[2], bF[2];
#pragma unroll
    for (int mt = 0; mt < 2; ++mt) {
      const _Float16* r = &As[waveM * 32 + mt * 16 + m16][0];
#pragma unroll
      for (int i = 0; i < 8; ++i)
        aF[mt].u[i] = *reinterpret_cast<const uint32_t*>(r + a_k_of(i, hi));
    }
#pragma unroll
    for (int nt = 0; nt < 2; ++nt) {
      const _Float16* r = &Bs[waveN * 32 + nt * 16 + m16][0];
#pragma unroll
      for (int i = 0; i < 8; ++i)
        bF[nt].u[i] = *reinterpret_cast<const uint32_t*>(r + hi * 16 + 2 * i);
    }
#pragma unroll
    for (int mt = 0; mt < 2; ++mt)
#pragma unroll
      for (int nt = 0; nt < 2; ++nt)
        acc[mt][nt] = __builtin_amdgcn_wmma_f32_16x16x32_f16(
            false, aF[mt].v, false, bF[nt].v, (short)0, acc[mt][nt], false, false);
    __syncthreads();
  }

  // --- epilogue ---
#pragma unroll
  for (int mt = 0; mt < 2; ++mt) {
#pragma unroll
    for (int nt = 0; nt < 2; ++nt) {
      const int gn = blockN * 64 + waveN * 32 + nt * 16 + m16;
      const float bv = bias[gn];
#pragma unroll
      for (int r = 0; r < 8; ++r) {
        const int gm = blockM * 128 + waveM * 32 + mt * 16 + hi * 8 + r;
        float v = acc[mt][nt][r] + bv;
        if (MODE == 0) {
          const int bb = gm >> 11, t = gm & 2047;
          const int head = gn >> 6, d = gn & 63;
          out16[(((size_t)(bb * 8 + head)) * 2048 + t) * 64 + d] = (_Float16)v;
        } else if (MODE == 2) {
          v = (v > 0.f) ? v : 0.01f * v;
          out32[(size_t)gm * 512 + gn] = v;
        } else {
          out32[(size_t)gm * 512 + gn] = v;
        }
      }
    }
  }
}

// ---------------------------------------------------------------------------
// Flash attention: 1 block per (b, h, 64-query tile); 4 waves, 16 rows/wave.
// dh = 64, causal (mask only on diagonal tile), softmax in exp2 domain.
// K tile async-DMA'd to LDS (shared by all 4 waves, merged b128 frag reads);
// V tile transposed into LDS with packed b32 stores.
// ---------------------------------------------------------------------------
__global__ __launch_bounds__(128) void attn_kernel(
    const _Float16* __restrict__ Qh, const _Float16* __restrict__ Kh,
    const _Float16* __restrict__ Vh, float* __restrict__ X) {
  __shared__ __align__(16) _Float16 Ks[64][72];      // [key][d], async raw rows
  __shared__ __align__(16) _Float16 Vt[64][68];      // [d][key], transposed
  __shared__ __align__(16) _Float16 Pl[4][16][72];   // per-wave P [row][col]

  const int qt = blockIdx.x;
  const int h  = blockIdx.y;
  const int b  = blockIdx.z;
  const int bh = b * 8 + h;
  const int tid  = threadIdx.x;
  const int lane = tid & 31;
  const int w    = tid >> 5;
  const int m16  = lane & 15, hi = lane >> 4;

  const size_t base = (size_t)bh * 2048 * 64;
  const _Float16* Qrow  = Qh + base + (size_t)(qt * 64 + w * 16 + m16) * 64;
  const _Float16* Kbase = Kh + base;
  const _Float16* Vbase = Vh + base;

  FragH aQ[2];
#pragma unroll
  for (int ks = 0; ks < 2; ++ks)
#pragma unroll
    for (int i = 0; i < 8; ++i)
      aQ[ks].u[i] = *reinterpret_cast<const uint32_t*>(Qrow + ks * 32 + a_k_of(i, hi));

  v8f zero = {};
  v8f o[4];
#pragma unroll
  for (int d = 0; d < 4; ++d) o[d] = zero;
  float mrow[8], lrow[8];
#pragma unroll
  for (int r = 0; r < 8; ++r) { mrow[r] = -__builtin_inff(); lrow[r] = 0.f; }

  // scores scaled by 1/sqrt(64) and moved to log2 domain in one multiply
  const float SCL = 0.125f * 1.44269504088896340736f;

  // K async staging: 16B x4 per thread (row = tid>>1, 32-half group = tid&1)
  const int kkey = tid >> 1;
  const int kc0  = (tid & 1) * 32;
  const uint32_t kLds0 = (uint32_t)(uintptr_t)&Ks[kkey][kc0];
  // V transpose staging: row pair (tid>>2)*2, 16 cols (tid&3)*16
  const int vkr = (tid >> 2) * 2;
  const int vc0 = (tid & 3) * 16;

  for (int kt = 0; kt <= qt; ++kt) {
    // ---- K tile: async DMA global -> LDS raw rows ----
    const _Float16* ksrc = Kbase + (size_t)(kt * 64 + kkey) * 64 + kc0;
#pragma unroll
    for (int jj = 0; jj < 4; ++jj)
      async_copy_b128(kLds0 + jj * 16, (uint64_t)(uintptr_t)(ksrc + jj * 8));
    // ---- V tile: transpose-stage Vt[d][key] with packed b32 stores ----
    const _Float16* v0p = Vbase + (size_t)(kt * 64 + vkr) * 64 + vc0;
    uint4 r0a = *reinterpret_cast<const uint4*>(v0p);
    uint4 r0b = *reinterpret_cast<const uint4*>(v0p + 8);
    uint4 r1a = *reinterpret_cast<const uint4*>(v0p + 64);
    uint4 r1b = *reinterpret_cast<const uint4*>(v0p + 72);
    const _Float16* h0a = reinterpret_cast<const _Float16*>(&r0a);
    const _Float16* h0b = reinterpret_cast<const _Float16*>(&r0b);
    const _Float16* h1a = reinterpret_cast<const _Float16*>(&r1a);
    const _Float16* h1b = reinterpret_cast<const _Float16*>(&r1b);
#pragma unroll
    for (int e = 0; e < 8; ++e) {
      PackH pa; pa.h[0] = h0a[e]; pa.h[1] = h1a[e];
      *reinterpret_cast<uint32_t*>(&Vt[vc0 + e][vkr]) = pa.u;
      PackH pb; pb.h[0] = h0b[e]; pb.h[1] = h1b[e];
      *reinterpret_cast<uint32_t*>(&Vt[vc0 + 8 + e][vkr]) = pb.u;
    }
    wait_asynccnt0();
    __syncthreads();

    // ---- S = Q @ K^T (K frags: merged b128 LDS reads, shared by 4 waves) ----
    v8f s[4];
#pragma unroll
    for (int nt = 0; nt < 4; ++nt) s[nt] = zero;
#pragma unroll
    for (int nt = 0; nt < 4; ++nt) {
      const _Float16* Krow = &Ks[nt * 16 + m16][0];
#pragma unroll
      for (int ks = 0; ks < 2; ++ks) {
        FragH bK;
#pragma unroll
        for (int i = 0; i < 8; ++i)
          bK.u[i] = *reinterpret_cast<const uint32_t*>(Krow + ks * 32 + hi * 16 + 2 * i);
        s[nt] = __builtin_amdgcn_wmma_f32_16x16x32_f16(
            false, aQ[ks].v, false, bK.v, (short)0, s[nt], false, false);
      }
    }
    // ---- scale (+ causal mask only on the diagonal tile) ----
    if (kt == qt) {
      const int qbase = qt * 64 + w * 16 + hi * 8;
#pragma unroll
      for (int nt = 0; nt < 4; ++nt) {
        const int key = kt * 64 + nt * 16 + m16;
#pragma unroll
        for (int r = 0; r < 8; ++r) {
          float sv = s[nt][r] * SCL;
          if (key > qbase + r) sv = -__builtin_inff();
          s[nt][r] = sv;
        }
      }
    } else {
#pragma unroll
      for (int nt = 0; nt < 4; ++nt)
#pragma unroll
        for (int r = 0; r < 8; ++r) s[nt][r] *= SCL;
    }
    // ---- online softmax stats (log2 domain; 16-lane-half row reductions) ----
    float alpha[8];
#pragma unroll
    for (int r = 0; r < 8; ++r) {
      float mx = s[0][r];
#pragma unroll
      for (int nt = 1; nt < 4; ++nt) mx = fmaxf(mx, s[nt][r]);
#pragma unroll
      for (int off = 8; off; off >>= 1) mx = fmaxf(mx, __shfl_xor(mx, off));
      const float mnew = fmaxf(mrow[r], mx);
      alpha[r] = exp2f(mrow[r] - mnew);
      float ssum = 0.f;
#pragma unroll
      for (int nt = 0; nt < 4; ++nt) {
        const float p = exp2f(s[nt][r] - mnew);
        s[nt][r] = p;  // reuse S registers as P
        ssum += p;
      }
#pragma unroll
      for (int off = 8; off; off >>= 1) ssum += __shfl_xor(ssum, off);
      lrow[r] = lrow[r] * alpha[r] + ssum;
      mrow[r] = mnew;
    }
    // ---- rescale O ----
#pragma unroll
    for (int d = 0; d < 4; ++d)
#pragma unroll
      for (int r = 0; r < 8; ++r) o[d][r] *= alpha[r];
    // ---- P: scatter u16 stores (cvt results), contiguous b128-merged reads ----
#pragma unroll
    for (int nt = 0; nt < 4; ++nt)
#pragma unroll
      for (int r = 0; r < 8; ++r)
        Pl[w][hi * 8 + r][nt * 16 + m16] = (_Float16)s[nt][r];
    FragH aP[2];
#pragma unroll
    for (int ks = 0; ks < 2; ++ks) {
      const _Float16* pr = &Pl[w][m16][0];
#pragma unroll
      for (int i = 0; i < 8; ++i)
        aP[ks].u[i] = *reinterpret_cast<const uint32_t*>(pr + ks * 32 + a_k_of(i, hi));
    }
    // ---- O += P @ V (V frags from transposed LDS tile, b32/b64 reads) ----
#pragma unroll
    for (int d = 0; d < 4; ++d) {
      const int dcol = d * 16 + m16;
#pragma unroll
      for (int ks = 0; ks < 2; ++ks) {
        FragH bV;
#pragma unroll
        for (int i = 0; i < 8; ++i) {
          const int k0 = ks * 32 + hi * 16 + 2 * i;
          bV.u[i] = *reinterpret_cast<const uint32_t*>(&Vt[dcol][k0]);
        }
        o[d] = __builtin_amdgcn_wmma_f32_16x16x32_f16(
            false, aP[ks].v, false, bV.v, (short)0, o[d], false, false);
      }
    }
    __syncthreads();  // Ks/Vt reused next iteration
  }

  // ---- finalize: O / l, write x[b,t,h*64+d] as f32 for Wo GEMM ----
  float inv[8];
#pragma unroll
  for (int r = 0; r < 8; ++r) inv[r] = 1.f / lrow[r];
#pragma unroll
  for (int d = 0; d < 4; ++d) {
    const int col = h * 64 + d * 16 + m16;
#pragma unroll
    for (int r = 0; r < 8; ++r) {
      const int gm = b * 2048 + qt * 64 + w * 16 + hi * 8 + r;
      X[(size_t)gm * 512 + col] = o[d][r] * inv[r];
    }
  }
}

// ---------------------------------------------------------------------------
// Fused residual-add + LayerNorm over 512 cols; one block per row.
// ---------------------------------------------------------------------------
__global__ __launch_bounds__(256) void ln_kernel(
    const float* __restrict__ x, const float* __restrict__ res,
    const float* __restrict__ g, const float* __restrict__ be,
    float* __restrict__ out) {
  const int row = blockIdx.x;
  const int t   = threadIdx.x;
  const float* xr = x + (size_t)row * 512;
  const float* rr = res + (size_t)row * 512;
  const float v0 = xr[t] + rr[t];
  const float v1 = xr[t + 256] + rr[t + 256];
  float s  = v0 + v1;
  float sq = v0 * v0 + v1 * v1;
#pragma unroll
  for (int off = 16; off; off >>= 1) {
    s  += __shfl_xor(s, off);
    sq += __shfl_xor(sq, off);
  }
  __shared__ float rs[8], rq[8];
  const int w = t >> 5, lane = t & 31;
  if (lane == 0) { rs[w] = s; rq[w] = sq; }
  __syncthreads();
  float ts = 0.f, tq = 0.f;
#pragma unroll
  for (int i = 0; i < 8; ++i) { ts += rs[i]; tq += rq[i]; }
  const float mean = ts * (1.f / 512.f);
  const float var  = tq * (1.f / 512.f) - mean * mean;
  const float inv  = rsqrtf(var + 1e-5f);
  out[(size_t)row * 512 + t]       = (v0 - mean) * inv * g[t] + be[t];
  out[(size_t)row * 512 + t + 256] = (v1 - mean) * inv * g[t + 256] + be[t + 256];
}

// ---------------------------------------------------------------------------
// Transpose + convert weights: Wt[n][k] = (f16) W[k][n]  (512x512, one-shot)
// ---------------------------------------------------------------------------
__global__ __launch_bounds__(256) void cvtT_f32_f16(const float* __restrict__ src,
                                                    _Float16* __restrict__ dst) {
  __shared__ float tile[32][33];
  const int bx = blockIdx.x * 32, by = blockIdx.y * 32;
  const int tx = threadIdx.x, ty = threadIdx.y;  // (32, 8)
#pragma unroll
  for (int j = 0; j < 32; j += 8)
    tile[ty + j][tx] = src[(size_t)(by + ty + j) * 512 + bx + tx];
  __syncthreads();
#pragma unroll
  for (int j = 0; j < 32; j += 8)
    dst[(size_t)(bx + ty + j) * 512 + by + tx] = (_Float16)tile[tx][ty + j];
}

// ---------------------------------------------------------------------------
extern "C" void kernel_launch(void* const* d_in, const int* in_sizes, int n_in,
                              void* d_out, int out_size, void* d_ws, size_t ws_size,
                              hipStream_t stream) {
  (void)in_sizes; (void)n_in; (void)out_size; (void)ws_size;
  const float* q  = (const float*)d_in[0];
  const float* k  = (const float*)d_in[1];
  const float* v  = (const float*)d_in[2];
  const float* Wq = (const float*)d_in[3];
  const float* bq = (const float*)d_in[4];
  const float* Wk = (const float*)d_in[5];
  const float* bk = (const float*)d_in[6];
  const float* Wv = (const float*)d_in[7];
  const float* bv = (const float*)d_in[8];
  const float* Wo = (const float*)d_in[9];
  const float* bo = (const float*)d_in[10];
  const float* Wh = (const float*)d_in[11];
  const float* bh = (const float*)d_in[12];
  const float* W2 = (const float*)d_in[13];
  const float* b2 = (const float*)d_in[14];
  const float* g1  = (const float*)d_in[15];
  const float* be1 = (const float*)d_in[16];
  const float* g2  = (const float*)d_in[17];
  const float* be2 = (const float*)d_in[18];

  char* ws = (char*)d_ws;
  size_t off = 0;
  auto alloc = [&](size_t bytes) -> void* {
    void* p = ws + off;
    off += (bytes + 255) & ~(size_t)255;
    return p;
  };

  const size_t wbytes = (size_t)512 * 512 * sizeof(_Float16);
  const size_t actH   = (size_t)8192 * 512 * sizeof(_Float16);
  const size_t actF   = (size_t)8192 * 512 * sizeof(float);

  _Float16* Wq16 = (_Float16*)alloc(wbytes);
  _Float16* Wk16 = (_Float16*)alloc(wbytes);
  _Float16* Wv16 = (_Float16*)alloc(wbytes);
  _Float16* Wo16 = (_Float16*)alloc(wbytes);
  _Float16* Wh16 = (_Float16*)alloc(wbytes);
  _Float16* W216 = (_Float16*)alloc(wbytes);
  _Float16* QhB  = (_Float16*)alloc(actH);
  _Float16* KhB  = (_Float16*)alloc(actH);
  _Float16* VhB  = (_Float16*)alloc(actH);
  float* Xat  = (float*)alloc(actF);
  float* Hbuf = (float*)alloc(actF);
  float* F1   = (float*)alloc(actF);
  float* Tmp  = (float*)alloc(actF);

  // weights -> f16, transposed to [n][k]
  const dim3 tg(16, 16), tb(32, 8);
  cvtT_f32_f16<<<tg, tb, 0, stream>>>(Wq, Wq16);
  cvtT_f32_f16<<<tg, tb, 0, stream>>>(Wk, Wk16);
  cvtT_f32_f16<<<tg, tb, 0, stream>>>(Wv, Wv16);
  cvtT_f32_f16<<<tg, tb, 0, stream>>>(Wo, Wo16);
  cvtT_f32_f16<<<tg, tb, 0, stream>>>(Wh, Wh16);
  cvtT_f32_f16<<<tg, tb, 0, stream>>>(W2, W216);

  const dim3 gg(8, 64), gb(256);
  // Q/K/V projections with fused bias + head permute (f16 out)
  gemm512<0><<<gg, gb, 0, stream>>>(q, Wq16, bq, QhB, nullptr);
  gemm512<0><<<gg, gb, 0, stream>>>(k, Wk16, bk, KhB, nullptr);
  gemm512<0><<<gg, gb, 0, stream>>>(v, Wv16, bv, VhB, nullptr);

  // flash attention -> x [B*T, 512] f32
  attn_kernel<<<dim3(32, 8, 4), 128, 0, stream>>>(QhB, KhB, VhB, Xat);

  // out proj, residual + LN1
  gemm512<1><<<gg, gb, 0, stream>>>(Xat, Wo16, bo, nullptr, Tmp);
  ln_kernel<<<8192, 256, 0, stream>>>(Tmp, q, g1, be1, Hbuf);

  // FFN with fused leaky_relu, then residual + LN2 -> d_out
  gemm512<2><<<gg, gb, 0, stream>>>(Hbuf, Wh16, bh, nullptr, F1);
  gemm512<1><<<gg, gb, 0, stream>>>(F1, W216, b2, nullptr, Tmp);
  ln_kernel<<<8192, 256, 0, stream>>>(Tmp, Hbuf, g2, be2, (float*)d_out);
}
